// EGNN_39298950759105
// MI455X (gfx1250) — matrix operand order repaired and checked by reading, not compiled, not run on hardware
//
#include <hip/hip_runtime.h>

// ---------------------------------------------------------------------------
// Types for CDNA5 WMMA (gfx1250, wave32)
// ---------------------------------------------------------------------------
typedef __attribute__((ext_vector_type(16))) __bf16        v16bf;
typedef __attribute__((ext_vector_type(8)))  float         v8f;
typedef __attribute__((ext_vector_type(8)))  unsigned int  v8u;

#define HID 128
#define NLAYERS 6
#define MT 4               // M-tiles (of 16 rows) per block
#define MROWS (MT * 16)    // 64 rows per block
// A-tile LDS row pitch in bf16 elements: 136 shorts = 272B = 17*16B.
// Odd multiple of 16B staggers consecutive rows by 4 banks -> the 16 rows of
// each ds_load_b128 cover banks 0-3,4-7,...,60-63 (conflict-free).
#define AP (HID + 8)

__device__ __forceinline__ unsigned short f2bf(float f) {
    unsigned int u = __float_as_uint(f);
    unsigned int r = (u + 0x7FFFu + ((u >> 16) & 1u)) >> 16;  // RNE
    return (unsigned short)r;
}
__device__ __forceinline__ unsigned int pack2bf(float a, float b) {
    return (unsigned int)f2bf(a) | ((unsigned int)f2bf(b) << 16);
}
__device__ __forceinline__ float rcp_f(float x) { return __builtin_amdgcn_rcpf(x); }
__device__ __forceinline__ float sigm_f(float x) { return rcp_f(1.f + __expf(-x)); }
__device__ __forceinline__ float silu_f(float x) { return x * sigm_f(x); }
__device__ __forceinline__ float tanh_f(float x) {
    x = fminf(fmaxf(x, -15.f), 15.f);
    float e = __expf(2.f * x);
    return (e - 1.f) * rcp_f(e + 1.f);
}

// ---------------------------------------------------------------------------
// Multi-tile WMMA: MT C-tiles (16x16,f32) += A-tiles(16x128 bf16, LDS) * B.
// B fragment loaded once per k-step and reused for all MT A-tiles.
// A layout per ISA 16-bit A-matrix: m=lane&15, half=lane>>4,
//   VGPR v holds K = ks*32 + 2*(v&3) + 16*(v>>2) + 8*half  (pair K,K+1)
//   -> 2x ds_load_b128 per tile per k-step (bank-conflict-free via AP pitch).
// Bpack layout: ((ks*8+ntile)*32 + lane)*8 + v -> 2x global_load_b128,
//   wave reads 1KB contiguous per fragment (L2-resident weights).
// ---------------------------------------------------------------------------
__device__ __forceinline__ void wmma4_mt(const unsigned short* As,   // [MROWS][AP]
                                         const unsigned int* __restrict__ Bp,
                                         int ntile, int lane, v8f* acc) {
    const int half = lane >> 4;
    const int mrow = lane & 15;
#pragma unroll
    for (int ks = 0; ks < 4; ++ks) {
        v8u b = *(const v8u*)(Bp + ((((ks * 8 + ntile) << 5) + lane) << 3));
#pragma unroll
        for (int t = 0; t < MT; ++t) {
            v8u a;
#pragma unroll
            for (int v = 0; v < 8; ++v) {
                int k = ks * 32 + ((v & 3) << 1) + ((v >> 2) << 4) + (half << 3);
                a[v] = *(const unsigned int*)(As + (t * 16 + mrow) * AP + k);
            }
            acc[t] = __builtin_amdgcn_wmma_f32_16x16x32_bf16(
                false, __builtin_bit_cast(v16bf, a),
                false, __builtin_bit_cast(v16bf, b),
                (short)0, acc[t], false, false);
        }
    }
}

// ---------------------------------------------------------------------------
// Pack one 128x128 f32 weight matrix into WMMA B-operand layout (bf16 pairs).
// dst index = ((ks*8 + ntile)*32 + lane)*8 + v
// ---------------------------------------------------------------------------
__global__ void pack_w_kernel(const float* __restrict__ src,
                              unsigned int* __restrict__ dst) {
    int idx = blockIdx.x * blockDim.x + threadIdx.x;
    if (idx >= 8192) return;
    int v = idx & 7;
    int r = idx >> 3;
    int lane = r & 31;  r >>= 5;
    int ntile = r & 7;
    int ks = r >> 3;
    int half = lane >> 4;
    int k = ks * 32 + ((v & 3) << 1) + ((v >> 2) << 4) + (half << 3);
    int n = (ntile << 4) + (lane & 15);
    dst[idx] = pack2bf(src[(size_t)k * HID + n], src[(size_t)(k + 1) * HID + n]);
}

__global__ void fill_kernel(float* p, long long cnt, float v) {
    long long i = (long long)blockIdx.x * blockDim.x + threadIdx.x;
    if (i < cnt) p[i] = v;
}

// ---------------------------------------------------------------------------
// Encoder stage 1: t = silu([scal,speed] @ enc_w1 + b1); also init xc.
// ---------------------------------------------------------------------------
__global__ void enc1_kernel(const float* __restrict__ x,
                            const float* __restrict__ w1,
                            const float* __restrict__ b1,
                            float* __restrict__ t, float* __restrict__ xc, int N_) {
    int idx = blockIdx.x * blockDim.x + threadIdx.x;
    int node = idx >> 7, c = idx & 127;
    if (node >= N_) return;
    float x0 = x[node * 3 + 0], x1 = x[node * 3 + 1], sc = x[node * 3 + 2];
    float sp = sqrtf(x0 * x0 + x1 * x1);
    float vv = sc * w1[c] + sp * w1[HID + c] + b1[c];
    t[(size_t)node * HID + c] = silu_f(vv);
    if (c == 0) { xc[node * 2 + 0] = x0; xc[node * 2 + 1] = x1; }
}

// ---------------------------------------------------------------------------
// Generic [M,128] x [128,128] WMMA GEMM, 64 rows/block, with epilogues.
// mode 0: out = u ; mode 1: out = silu(u) ; mode 2: out = layernorm(u)*g+beta
// ---------------------------------------------------------------------------
__global__ __launch_bounds__(256)
void gemm128_kernel(const float* __restrict__ A, const unsigned int* __restrict__ Bp,
                    const float* __restrict__ bias, float* __restrict__ out,
                    int M, int mode,
                    const float* __restrict__ gamma, const float* __restrict__ beta) {
    __shared__ unsigned short As[MROWS * AP];
    __shared__ float U[MROWS * HID];             // 32KB (mode 2 only)
    int tid = threadIdx.x;
    int r0 = blockIdx.x * MROWS;
    for (int i = tid; i < MROWS * 32; i += 256) {
        int m = i >> 5, c4 = (i & 31) << 2;
        int rr = r0 + m; if (rr > M - 1) rr = M - 1;
        float4 a = *(const float4*)(A + (size_t)rr * HID + c4);
        *(uint2*)(As + m * AP + c4) = make_uint2(pack2bf(a.x, a.y), pack2bf(a.z, a.w));
    }
    __syncthreads();
    int lane = tid & 31, ntile = tid >> 5, half = lane >> 4, nloc = lane & 15;
    int n = (ntile << 4) + nloc;
    v8f acc[MT];
#pragma unroll
    for (int t = 0; t < MT; ++t) acc[t] = (v8f){0.f,0.f,0.f,0.f,0.f,0.f,0.f,0.f};
    wmma4_mt(As, Bp, ntile, lane, acc);
    float bb = bias[n];
#pragma unroll
    for (int t = 0; t < MT; ++t) {
#pragma unroll
        for (int i = 0; i < 8; ++i) {
            int row = t * 16 + i + (half << 3);
            float u = acc[t][i] + bb;
            if (mode == 1) u = silu_f(u);
            if (mode == 2) {
                U[row * HID + n] = u;
            } else {
                int rr = r0 + row;
                if (rr < M) out[(size_t)rr * HID + n] = u;
            }
        }
    }
    if (mode == 2) {
        __syncthreads();
        // 16 lanes per row, shuffle-segment reductions, 4 row-groups
#pragma unroll
        for (int rp = 0; rp < MT; ++rp) {
            int r = rp * 16 + (tid >> 4), j = tid & 15, c0 = j << 3;
            float lv[8];
            float s = 0.f;
#pragma unroll
            for (int q = 0; q < 8; ++q) { lv[q] = U[r * HID + c0 + q]; s += lv[q]; }
            s += __shfl_down(s, 8, 16); s += __shfl_down(s, 4, 16);
            s += __shfl_down(s, 2, 16); s += __shfl_down(s, 1, 16);
            float mean = __shfl(s, 0, 16) * (1.f / HID);
            float vs = 0.f;
#pragma unroll
            for (int q = 0; q < 8; ++q) { float d = lv[q] - mean; vs += d * d; }
            vs += __shfl_down(vs, 8, 16); vs += __shfl_down(vs, 4, 16);
            vs += __shfl_down(vs, 2, 16); vs += __shfl_down(vs, 1, 16);
            float rstd = rsqrtf(__shfl(vs, 0, 16) * (1.f / HID) + 1e-5f);
            int rr = r0 + r;
            if (rr < M) {
#pragma unroll
                for (int q = 0; q < 8; ++q) {
                    int c = c0 + q;
                    out[(size_t)rr * HID + c] = (lv[q] - mean) * rstd * gamma[c] + beta[c];
                }
            }
        }
    }
}

// ---------------------------------------------------------------------------
// Edge kernel: 64 edges/block, 8 waves. Fused edge-MLP (2 WMMA GEMM stages
// with 4-way B-fragment reuse), shuffle LN + gating, atomic scatter.
// LDS: [hsb|hdb] (34KB) aliased by m2f (32KB) after stage 1; m1b separate.
// ---------------------------------------------------------------------------
__global__ __launch_bounds__(256)
void edge_kernel(const float* __restrict__ h, const float* __restrict__ xc,
                 const long long* __restrict__ ei, int E_,
                 const unsigned int* __restrict__ w1a, const unsigned int* __restrict__ w1b,
                 const float* __restrict__ w1c, const float* __restrict__ b1,
                 const unsigned int* __restrict__ w2, const float* __restrict__ b2,
                 const float* __restrict__ eg, const float* __restrict__ ebn,
                 const float* __restrict__ gw, const float* __restrict__ gb,
                 const float* __restrict__ xw, const float* __restrict__ xb,
                 float* __restrict__ agg_h, float* __restrict__ agg_x,
                 float* __restrict__ deg) {
    __shared__ __align__(16) unsigned char smem[MROWS * AP * 6];   // 51KB
    unsigned short* hsb = (unsigned short*)smem;                   // 17KB [64][AP]
    unsigned short* hdb = hsb + MROWS * AP;                        // 17KB
    float*          m2f = (float*)smem;                            // 32KB (aliases hsb+hdb)
    unsigned short* m1b = (unsigned short*)(smem + MROWS * AP * 4); // 17KB
    __shared__ int   sS[MROWS], sD[MROWS];
    __shared__ float sDsq[MROWS], sRx[MROWS], sRy[MROWS];

    int tid = threadIdx.x;
    int e0 = blockIdx.x * MROWS;
    if (tid < MROWS) {
        int e = e0 + tid;
        int ec = (e < E_) ? e : 0;
        int s = (int)ei[ec];
        int d = (int)ei[(long long)E_ + ec];
        sS[tid] = s; sD[tid] = d;
        float rx = xc[s * 2 + 0] - xc[d * 2 + 0];
        float ry = xc[s * 2 + 1] - xc[d * 2 + 1];
        float dsq = rx * rx + ry * ry;
        float inv = rcp_f(sqrtf(dsq) + 1e-8f);
        sDsq[tid] = dsq; sRx[tid] = rx * inv; sRy[tid] = ry * inv;
    }
    __syncthreads();
    for (int i = tid; i < MROWS * 32; i += 256) {
        int m = i >> 5, c4 = (i & 31) << 2;
        float4 a = *(const float4*)(h + (size_t)sS[m] * HID + c4);
        float4 b = *(const float4*)(h + (size_t)sD[m] * HID + c4);
        *(uint2*)(hsb + m * AP + c4) = make_uint2(pack2bf(a.x, a.y), pack2bf(a.z, a.w));
        *(uint2*)(hdb + m * AP + c4) = make_uint2(pack2bf(b.x, b.y), pack2bf(b.z, b.w));
    }
    __syncthreads();

    int lane = tid & 31, ntile = tid >> 5, half = lane >> 4, nloc = lane & 15;
    int n = (ntile << 4) + nloc;

    // stage 1: m1 = silu(h_s @ w1a + h_d @ w1b + dsq * w1c + b1)
    v8f acc[MT];
#pragma unroll
    for (int t = 0; t < MT; ++t) acc[t] = (v8f){0.f,0.f,0.f,0.f,0.f,0.f,0.f,0.f};
    wmma4_mt(hsb, w1a, ntile, lane, acc);
    wmma4_mt(hdb, w1b, ntile, lane, acc);
    float w1cn = w1c[n], b1n = b1[n];
#pragma unroll
    for (int t = 0; t < MT; ++t) {
#pragma unroll
        for (int i = 0; i < 8; ++i) {
            int row = t * 16 + i + (half << 3);
            float u = acc[t][i] + sDsq[row] * w1cn + b1n;
            m1b[row * AP + n] = f2bf(silu_f(u));
        }
    }
    __syncthreads();   // all stage-1 reads of hsb/hdb done; m1b complete

    // stage 2: m2 = silu(m1 @ w2 + b2)   (m2f aliases hsb/hdb - safe now)
    v8f acc2[MT];
#pragma unroll
    for (int t = 0; t < MT; ++t) acc2[t] = (v8f){0.f,0.f,0.f,0.f,0.f,0.f,0.f,0.f};
    wmma4_mt(m1b, w2, ntile, lane, acc2);
    float b2n = b2[n];
#pragma unroll
    for (int t = 0; t < MT; ++t) {
#pragma unroll
        for (int i = 0; i < 8; ++i) {
            int row = t * 16 + i + (half << 3);
            m2f[row * HID + n] = silu_f(acc2[t][i] + b2n);
        }
    }
    __syncthreads();

    // row-parallel LN + gate + coord weight + scatter: 16 lanes per row
#pragma unroll
    for (int rp = 0; rp < MT; ++rp) {
        int r = rp * 16 + (tid >> 4);   // edge row 0..63
        int j = tid & 15;
        int c0 = j << 3;
        float lv[8];
        float s = 0.f;
#pragma unroll
        for (int q = 0; q < 8; ++q) { lv[q] = m2f[r * HID + c0 + q]; s += lv[q]; }
        s += __shfl_down(s, 8, 16); s += __shfl_down(s, 4, 16);
        s += __shfl_down(s, 2, 16); s += __shfl_down(s, 1, 16);
        float mean = __shfl(s, 0, 16) * (1.f / HID);
        float vs = 0.f;
#pragma unroll
        for (int q = 0; q < 8; ++q) { float d = lv[q] - mean; vs += d * d; }
        vs += __shfl_down(vs, 8, 16); vs += __shfl_down(vs, 4, 16);
        vs += __shfl_down(vs, 2, 16); vs += __shfl_down(vs, 1, 16);
        float rstd = rsqrtf(__shfl(vs, 0, 16) * (1.f / HID) + 1e-5f);
        float ga = 0.f;
#pragma unroll
        for (int q = 0; q < 8; ++q) {
            int c = c0 + q;
            lv[q] = (lv[q] - mean) * rstd * eg[c] + ebn[c];   // layernormed m
            ga += lv[q] * gw[c];
        }
        ga += __shfl_down(ga, 8, 16); ga += __shfl_down(ga, 4, 16);
        ga += __shfl_down(ga, 2, 16); ga += __shfl_down(ga, 1, 16);
        float alpha = sigm_f(__shfl(ga, 0, 16) + gb[0]);
        float xa = 0.f;
#pragma unroll
        for (int q = 0; q < 8; ++q) xa += lv[q] * alpha * xw[c0 + q];
        xa += __shfl_down(xa, 8, 16); xa += __shfl_down(xa, 4, 16);
        xa += __shfl_down(xa, 2, 16); xa += __shfl_down(xa, 1, 16);
        float cw = tanh_f(__shfl(xa, 0, 16) + xb[0]);

        if (e0 + r < E_) {
            int d = sD[r];
#pragma unroll
            for (int q = 0; q < 8; ++q)
                atomicAdd(&agg_h[(size_t)d * HID + c0 + q], lv[q] * alpha);
            if (j == 0) {
                atomicAdd(&agg_x[d * 2 + 0], sRx[r] * cw);
                atomicAdd(&agg_x[d * 2 + 1], sRy[r] * cw);
                atomicAdd(&deg[d], 1.0f);
            }
        }
    }
}

// ---------------------------------------------------------------------------
// Node update: h += silu([h, agg_h] @ hw1 + b1) @ hw2 + b2   (64 nodes/block)
// ---------------------------------------------------------------------------
__global__ __launch_bounds__(256)
void node_kernel(float* __restrict__ h, const float* __restrict__ agg_h, int N_,
                 const unsigned int* __restrict__ hw1a, const unsigned int* __restrict__ hw1b,
                 const float* __restrict__ b1,
                 const unsigned int* __restrict__ hw2, const float* __restrict__ b2) {
    __shared__ unsigned short A1[MROWS * AP];
    __shared__ unsigned short A2[MROWS * AP];
    __shared__ unsigned short T[MROWS * AP];
    int tid = threadIdx.x;
    int r0 = blockIdx.x * MROWS;
    for (int i = tid; i < MROWS * 32; i += 256) {
        int m = i >> 5, c4 = (i & 31) << 2;
        int rr = r0 + m; if (rr > N_ - 1) rr = N_ - 1;
        float4 a = *(const float4*)(h + (size_t)rr * HID + c4);
        float4 b = *(const float4*)(agg_h + (size_t)rr * HID + c4);
        *(uint2*)(A1 + m * AP + c4) = make_uint2(pack2bf(a.x, a.y), pack2bf(a.z, a.w));
        *(uint2*)(A2 + m * AP + c4) = make_uint2(pack2bf(b.x, b.y), pack2bf(b.z, b.w));
    }
    __syncthreads();
    int lane = tid & 31, ntile = tid >> 5, half = lane >> 4, nloc = lane & 15;
    int n = (ntile << 4) + nloc;
    v8f acc[MT];
#pragma unroll
    for (int t = 0; t < MT; ++t) acc[t] = (v8f){0.f,0.f,0.f,0.f,0.f,0.f,0.f,0.f};
    wmma4_mt(A1, hw1a, ntile, lane, acc);
    wmma4_mt(A2, hw1b, ntile, lane, acc);
    float b1n = b1[n];
#pragma unroll
    for (int t = 0; t < MT; ++t) {
#pragma unroll
        for (int i = 0; i < 8; ++i) {
            int row = t * 16 + i + (half << 3);
            T[row * AP + n] = f2bf(silu_f(acc[t][i] + b1n));
        }
    }
    __syncthreads();
    v8f acc2[MT];
#pragma unroll
    for (int t = 0; t < MT; ++t) acc2[t] = (v8f){0.f,0.f,0.f,0.f,0.f,0.f,0.f,0.f};
    wmma4_mt(T, hw2, ntile, lane, acc2);
    float b2n = b2[n];
#pragma unroll
    for (int t = 0; t < MT; ++t) {
#pragma unroll
        for (int i = 0; i < 8; ++i) {
            int row = t * 16 + i + (half << 3);
            int rr = r0 + row;
            if (rr < N_) h[(size_t)rr * HID + n] += acc2[t][i] + b2n;
        }
    }
}

// ---------------------------------------------------------------------------
// Coordinate update: xc += agg_x / max(deg, 1)
// ---------------------------------------------------------------------------
__global__ void coord_kernel(float* __restrict__ xc, const float* __restrict__ agg_x,
                             const float* __restrict__ deg, int N_) {
    int idx = blockIdx.x * blockDim.x + threadIdx.x;
    if (idx >= N_ * 2) return;
    float d = deg[idx >> 1];
    d = d > 1.f ? d : 1.f;
    xc[idx] += agg_x[idx] * rcp_f(d);
}

// ---------------------------------------------------------------------------
// Decoder magnitude: mag = t1 @ dm_w2 + dm_b2   (128 -> 1)
// ---------------------------------------------------------------------------
__global__ void mag_kernel(const float* __restrict__ t1, const float* __restrict__ w,
                           const float* __restrict__ b, float* __restrict__ mag, int N_) {
    int n = blockIdx.x * blockDim.x + threadIdx.x;
    if (n >= N_) return;
    float s = 0.f;
    for (int c = 0; c < HID; ++c) s += t1[(size_t)n * HID + c] * w[c];
    mag[n] = s + b[0];
}

// ---------------------------------------------------------------------------
// Final mix: out = sigmoid(mix)*(mag*dirn) + (1-sigmoid(mix))*res
// ---------------------------------------------------------------------------
__global__ void final_kernel(const float* __restrict__ t2, const float* __restrict__ drw2,
                             const float* __restrict__ drb2, const float* __restrict__ mag,
                             const float* __restrict__ xc, const float* __restrict__ mixp,
                             float* __restrict__ out, int N_) {
    int n = blockIdx.x * blockDim.x + threadIdx.x;
    if (n >= N_) return;
    float r0 = 0.f, r1 = 0.f;
    for (int c = 0; c < HID; ++c) {
        float tv = t2[(size_t)n * HID + c];
        r0 += tv * drw2[c * 2 + 0];
        r1 += tv * drw2[c * 2 + 1];
    }
    r0 += drb2[0]; r1 += drb2[1];
    float x0 = xc[n * 2 + 0], x1 = xc[n * 2 + 1];
    float nrm = sqrtf(x0 * x0 + x1 * x1);
    nrm = nrm > 1e-12f ? nrm : 1e-12f;
    float w = sigm_f(mixp[0]);
    float m = mag[n];
    out[n * 2 + 0] = w * m * (x0 / nrm) + (1.f - w) * r0;
    out[n * 2 + 1] = w * m * (x1 / nrm) + (1.f - w) * r1;
}

// ---------------------------------------------------------------------------
// Host orchestration
// ---------------------------------------------------------------------------
extern "C" void kernel_launch(void* const* d_in, const int* in_sizes, int n_in,
                              void* d_out, int out_size, void* d_ws, size_t ws_size,
                              hipStream_t stream) {
    const float*     x      = (const float*)d_in[0];
    const long long* ei     = (const long long*)d_in[1];
    const float*     enc_w1 = (const float*)d_in[2];
    const float*     enc_b1 = (const float*)d_in[3];
    const float*     enc_w2 = (const float*)d_in[4];
    const float*     enc_b2 = (const float*)d_in[5];
    const float*     enc_g  = (const float*)d_in[6];
    const float*     enc_bn = (const float*)d_in[7];
    const float*     e_w1   = (const float*)d_in[8];
    const float*     e_b1   = (const float*)d_in[9];
    const float*     e_w2   = (const float*)d_in[10];
    const float*     e_b2   = (const float*)d_in[11];
    const float*     e_g    = (const float*)d_in[12];
    const float*     e_bn   = (const float*)d_in[13];
    const float*     g_w    = (const float*)d_in[14];
    const float*     g_b    = (const float*)d_in[15];
    const float*     x_w    = (const float*)d_in[16];
    const float*     x_b    = (const float*)d_in[17];
    const float*     h_w1   = (const float*)d_in[18];
    const float*     h_b1   = (const float*)d_in[19];
    const float*     h_w2   = (const float*)d_in[20];
    const float*     h_b2   = (const float*)d_in[21];
    const float*     dm_w1  = (const float*)d_in[22];
    const float*     dm_b1  = (const float*)d_in[23];
    const float*     dm_w2  = (const float*)d_in[24];
    const float*     dm_b2  = (const float*)d_in[25];
    const float*     dr_w1  = (const float*)d_in[26];
    const float*     dr_b1  = (const float*)d_in[27];
    const float*     dr_w2  = (const float*)d_in[28];
    const float*     dr_b2  = (const float*)d_in[29];
    const float*     mix    = (const float*)d_in[30];

    const int N = in_sizes[0] / 3;
    const int E = in_sizes[1] / 2;

    float* ws = (float*)d_ws;
    size_t off = 0;
    float* h     = ws + off; off += (size_t)N * HID;
    float* xc    = ws + off; off += (size_t)N * 2;
    float* agg_h = ws + off; off += (size_t)N * HID;   // agg_h, agg_x, deg contiguous
    float* agg_x = ws + off; off += (size_t)N * 2;
    float* deg   = ws + off; off += (size_t)N;
    float* t1    = ws + off; off += (size_t)N * HID;
    float* mag   = ws + off; off += (size_t)N;
    unsigned int* pk = (unsigned int*)(ws + off);      // 39 * 8192 u32

    auto P = [&](int i) { return pk + (size_t)i * 8192; };
    auto packmat = [&](const float* src, unsigned int* dst) {
        pack_w_kernel<<<32, 256, 0, stream>>>(src, dst);
    };

    // ---- pack all 128x128 weight blocks into WMMA B layout (bf16) ----
    packmat(enc_w2, P(0));
    for (int l = 0; l < NLAYERS; ++l) {
        const float* w1 = e_w1 + (size_t)l * 257 * HID;
        packmat(w1,                       P(1 + l * 6 + 0));  // w1a (h_src rows 0..127)
        packmat(w1 + (size_t)128 * HID,   P(1 + l * 6 + 1));  // w1b (h_dst rows 128..255)
        packmat(e_w2 + (size_t)l * HID * HID, P(1 + l * 6 + 2));
        const float* hw1 = h_w1 + (size_t)l * 256 * HID;
        packmat(hw1,                      P(1 + l * 6 + 3));  // hw1a (h part)
        packmat(hw1 + (size_t)128 * HID,  P(1 + l * 6 + 4));  // hw1b (agg_h part)
        packmat(h_w2 + (size_t)l * HID * HID, P(1 + l * 6 + 5));
    }
    packmat(dm_w1, P(37));
    packmat(dr_w1, P(38));

    const int nodeBlocks = (N + MROWS - 1) / MROWS;
    const int edgeBlocks = (E + MROWS - 1) / MROWS;

    // ---- encoder ----
    enc1_kernel<<<((size_t)N * HID + 255) / 256, 256, 0, stream>>>(x, enc_w1, enc_b1, t1, xc, N);
    gemm128_kernel<<<nodeBlocks, 256, 0, stream>>>(t1, P(0), enc_b2, h, N, 2, enc_g, enc_bn);

    // ---- EGNN layers ----
    for (int l = 0; l < NLAYERS; ++l) {
        long long aggCnt = (long long)N * (HID + 2 + 1);
        fill_kernel<<<(unsigned)((aggCnt + 255) / 256), 256, 0, stream>>>(agg_h, aggCnt, 0.f);
        edge_kernel<<<edgeBlocks, 256, 0, stream>>>(
            h, xc, ei, E,
            P(1 + l * 6 + 0), P(1 + l * 6 + 1),
            e_w1 + (size_t)l * 257 * HID + (size_t)256 * HID,  // w1c = row 256
            e_b1 + (size_t)l * HID,
            P(1 + l * 6 + 2), e_b2 + (size_t)l * HID,
            e_g + (size_t)l * HID, e_bn + (size_t)l * HID,
            g_w + (size_t)l * HID, g_b + l,
            x_w + (size_t)l * HID, x_b + l,
            agg_h, agg_x, deg);
        node_kernel<<<nodeBlocks, 256, 0, stream>>>(
            h, agg_h, N,
            P(1 + l * 6 + 3), P(1 + l * 6 + 4), h_b1 + (size_t)l * HID,
            P(1 + l * 6 + 5), h_b2 + (size_t)l * HID);
        coord_kernel<<<(N * 2 + 255) / 256, 256, 0, stream>>>(xc, agg_x, deg, N);
    }

    // ---- decoders ----
    gemm128_kernel<<<nodeBlocks, 256, 0, stream>>>(h, P(37), dm_b1, t1, N, 1, nullptr, nullptr);
    mag_kernel<<<(N + 255) / 256, 256, 0, stream>>>(t1, dm_w2, dm_b2, mag, N);
    gemm128_kernel<<<nodeBlocks, 256, 0, stream>>>(h, P(38), dr_b1, t1, N, 1, nullptr, nullptr);
    final_kernel<<<(N + 255) / 256, 256, 0, stream>>>(t1, dr_w2, dr_b2, mag, xc, mix,
                                                      (float*)d_out, N);
}
